// EnhancedQuantumLLM_45286135169503
// MI455X (gfx1250) — matrix operand order repaired and checked by reading, not compiled, not run on hardware
//
#include <hip/hip_runtime.h>
#include <math.h>

// ---------------------------------------------------------------------------
// EnhancedQuantumLLM attention for gfx1250 (MI455X, wave32, WMMA)
//
// Key algebraic identity: scale pattern is [H,L,1] (same across d), so
//   score_lm = p_l * p_m * S_lm,  S = (Qr+iQi)(Kr+iKi)^T  (scale-independent)
//   mag_lm   = ap[l] * ap[m] * |S_lm| / sqrt(D)
// => attention is a 4-temperature flash-softmax over one shared complex score.
// ---------------------------------------------------------------------------

typedef _Float16 h8   __attribute__((ext_vector_type(8)));
typedef _Float16 v16h __attribute__((ext_vector_type(16)));
typedef float    v8f  __attribute__((ext_vector_type(8)));

#define BDIM  2
#define HDIM  16
#define LDIM  1024
#define DDIM  64
#define LMAXI 2048
#define QTILE 64
#define KTILE 32
#define NWAVE 4
#define PI_F  3.14159265358979323846f

__device__ __forceinline__ v8f zero8() {
  v8f z;
#pragma unroll
  for (int j = 0; j < 8; ++j) z[j] = 0.f;
  return z;
}

__device__ __forceinline__ v16h mk16(h8 lo, h8 hi) {
  v16h r;
#pragma unroll
  for (int j = 0; j < 8; ++j) { r[j] = lo[j]; r[j + 8] = hi[j]; }
  return r;
}

__device__ __forceinline__ v8f wmma_f16(v16h a, v16h b, v8f c) {
  // D = A(16x32 f16) * B(32x16 f16) + C(16x16 f32)
  return __builtin_amdgcn_wmma_f32_16x16x32_f16(false, a, false, b,
                                                (short)0, c, false, false);
}

// A-fragment (16x32 f16) of Q from global f32 row:
// lane: M = lane&15 (row fixed); elems 0..7 -> d = dwin*32 + half*8 + e,
// elems 8..15 -> +16  (ISA 7.12.2, 16-bit A 16x32 layout)
__device__ __forceinline__ v16h ldA_q(const float* __restrict__ rowp,
                                      int dwin, int half) {
  const float* p0 = rowp + dwin * 32 + half * 8;
  v16h r;
#pragma unroll
  for (int j = 0; j < 8; ++j) {
    r[j]     = (_Float16)p0[j];
    r[j + 8] = (_Float16)p0[j + 16];
  }
  return r;
}

// B-fragment (32x16 f16) for scores from LDS K tile staged [key][d]:
// lane: N(col)=key = sub*16 + (lane&15); K-dim = d: elems 0..7 -> d half*8+e,
// elems 8..15 -> +16 within the dwin*32 window (matches A layout).
__device__ __forceinline__ v16h ldB_k(const _Float16* s, int sub, int dwin,
                                      int col, int half) {
  const _Float16* base = s + (sub * 16 + col) * DDIM + dwin * 32 + half * 8;
  return mk16(*(const h8*)(base), *(const h8*)(base + 16));
}

// B-fragment (32x16 f16) for PV from LDS V tile staged TRANSPOSED [d][key]:
// lane: N = d = dt*16 + (lane&15); K-dim = key: half*8+e, then +16 (A layout).
__device__ __forceinline__ v16h ldB_v(const _Float16* s, int dt,
                                      int col, int half) {
  const _Float16* base = s + (dt * 16 + col) * KTILE + half * 8;
  return mk16(*(const h8*)(base), *(const h8*)(base + 16));
}

// A-fragment (16x32 f16) of softmax weights from wave-private LDS [row][key]
__device__ __forceinline__ v16h ldA_p(const _Float16* s, int col, int half) {
  const _Float16* b0 = s + col * KTILE + half * 8;
  return mk16(*(const h8*)(b0), *(const h8*)(b0 + 16));
}

// ---------------------------------------------------------------------------
// Kernel 1: amplitude table ap[s][h][t] = |p_sh(t)| / ||p_sh||_LMAX
// ---------------------------------------------------------------------------
__global__ void ap_kernel(float* __restrict__ ap) {
  const int s = blockIdx.x >> 4;
  const int h = blockIdx.x & 15;
  const float freqs[4] = {1.0f, 0.5f, 0.25f, 0.1f};
  const float f    = freqs[s];
  const float ph   = 2.0f * PI_F * (float)h / (float)HDIM;
  const float step = 2.0f * PI_F * f / (float)(LMAXI - 1);  // linspace incl. end

  __shared__ float red[256];
  const int tid = threadIdx.x;
  float part = 0.f;
  for (int t = tid; t < LMAXI; t += 256) {
    float tt = step * (float)t;
    float pr = __cosf(tt + ph) + __cosf(2.f * tt + ph) + __cosf(0.5f * tt + ph);
    float pi = __sinf(tt + ph) + __sinf(2.f * tt + ph) + __sinf(0.5f * tt + ph);
    part += pr * pr + pi * pi;
  }
  red[tid] = part;
  __syncthreads();
  for (int o = 128; o > 0; o >>= 1) {
    if (tid < o) red[tid] += red[tid + o];
    __syncthreads();
  }
  const float rn = 1.0f / sqrtf(red[0]);
  float* dst = ap + (size_t)(s * HDIM + h) * LDIM;
  for (int t = tid; t < LDIM; t += 256) {
    float tt = step * (float)t;
    float pr = __cosf(tt + ph) + __cosf(2.f * tt + ph) + __cosf(0.5f * tt + ph);
    float pi = __sinf(tt + ph) + __sinf(2.f * tt + ph) + __sinf(0.5f * tt + ph);
    dst[t] = sqrtf(pr * pr + pi * pi) * rn;
  }
}

// ---------------------------------------------------------------------------
// Kernel 2: expert pattern [L,D], with 1/sqrt(4) scale-average and
// 1/(sqrt(3)*sqrt(8)*sqrt(LMAX)) normalization folded in.
// ---------------------------------------------------------------------------
__global__ void expert_kernel(float* __restrict__ epr, float* __restrict__ epi) {
  const int idx = blockIdx.x * 256 + threadIdx.x;   // < L*D = 65536
  const int t = idx >> 6, d = idx & 63;
  const float tt  = 2.0f * PI_F * (float)t / (float)(LMAXI - 1);
  const float phd = 2.0f * PI_F * (float)d / (float)DDIM;
  float cr = 0.f, ci = 0.f;
#pragma unroll
  for (int i = 0; i < 8; ++i) {
#pragma unroll
    for (int j = 0; j < 3; ++j) {
      float fq = (0.3f - 0.1f * (float)j) + 0.1f * (float)i;
      float a  = fq * tt + phd;
      cr += __cosf(a);
      ci += __sinf(a);
    }
  }
  const float scl = 0.5f / (sqrtf((float)LMAXI) * sqrtf(24.0f));
  epr[idx] = cr * scl;
  epi[idx] = ci * scl;
}

// ---------------------------------------------------------------------------
// Kernel 3: fused multi-scale flash attention + expert complex multiply.
// Grid: (L/QTILE, B*H). Block: 128 threads = 4 waves; wave w owns q-rows
// [qblk*64 + w*16, +16). Scale loop outermost (register economy), key tiles
// of 32 streamed through LDS, WMMA f16->f32 for both QK^T and PV.
// ---------------------------------------------------------------------------
__global__ __launch_bounds__(128)
void attn_kernel(const float* __restrict__ Qr, const float* __restrict__ Qi,
                 const float* __restrict__ Kr, const float* __restrict__ Ki,
                 const float* __restrict__ Vr, const float* __restrict__ Vi,
                 const float* __restrict__ ap,
                 const float* __restrict__ epr, const float* __restrict__ epi,
                 float* __restrict__ out) {
  __shared__ _Float16 s_kr[KTILE * DDIM];          // [key][d]
  __shared__ _Float16 s_ki[KTILE * DDIM];          // [key][d]
  __shared__ _Float16 s_vtr[DDIM * KTILE];         // [d][key] (transposed)
  __shared__ _Float16 s_vti[DDIM * KTILE];         // [d][key]
  __shared__ _Float16 s_p[NWAVE][16 * KTILE];      // wave-private W tiles
  __shared__ float    s_accR[QTILE * DDIM];        // cross-scale accumulators
  __shared__ float    s_accI[QTILE * DDIM];

  const int qblk = blockIdx.x;
  const int bh   = blockIdx.y;              // b*H + h
  const int h    = bh & (HDIM - 1);
  const int tid  = threadIdx.x;
  const int w    = tid >> 5;
  const int lane = tid & 31;
  const int col  = lane & 15;
  const int half = lane >> 4;
  const int qr0  = qblk * QTILE + w * 16;

  for (int i = tid; i < QTILE * DDIM; i += 128) { s_accR[i] = 0.f; s_accI[i] = 0.f; }

  // Q A-fragments (f16), resident for the whole kernel.
  const float* qrow_r = Qr + ((size_t)bh * LDIM + qr0 + col) * DDIM;
  const float* qrow_i = Qi + ((size_t)bh * LDIM + qr0 + col) * DDIM;
  v16h qA_r[2], qA_i[2];
#pragma unroll
  for (int dwin = 0; dwin < 2; ++dwin) {
    qA_r[dwin] = ldA_q(qrow_r, dwin, half);
    qA_i[dwin] = ldA_q(qrow_i, dwin, half);
  }

  const size_t kvbase0 = (size_t)bh * LDIM * DDIM;

  for (int s = 0; s < 4; ++s) {
    const float* apS = ap + (size_t)(s * HDIM + h) * LDIM;
    float apq[8];
#pragma unroll
    for (int r = 0; r < 8; ++r) apq[r] = apS[qr0 + r + 8 * half];

    float mrow[8], lrow[8];
#pragma unroll
    for (int r = 0; r < 8; ++r) { mrow[r] = -1e30f; lrow[r] = 0.f; }
    v8f oR[4], oI[4];
#pragma unroll
    for (int dt = 0; dt < 4; ++dt) { oR[dt] = zero8(); oI[dt] = zero8(); }

    for (int kt = 0; kt < LDIM / KTILE; ++kt) {
      __syncthreads();
      // ---- cooperative staging: K row-major, V transposed, both f16 ----
      {
        const size_t g = kvbase0 + (size_t)kt * KTILE * DDIM;
        for (int i = tid; i < KTILE * DDIM; i += 128) {
          int key = i >> 6, d = i & 63;
          s_kr[i] = (_Float16)Kr[g + i];
          s_ki[i] = (_Float16)Ki[g + i];
          s_vtr[d * KTILE + key] = (_Float16)Vr[g + i];
          s_vti[d * KTILE + key] = (_Float16)Vi[g + i];
        }
      }
      __syncthreads();

      // ---- complex scores S = (Qr+iQi)(Kr+iKi)^T over D=64 (16 WMMAs) ----
      v8f srt[2], sit[2];
#pragma unroll
      for (int sub = 0; sub < 2; ++sub) {
        v8f a1 = zero8(), a2 = zero8(), b1 = zero8();
#pragma unroll
        for (int dwin = 0; dwin < 2; ++dwin) {
          a1 = wmma_f16(qA_r[dwin], ldB_k(s_kr, sub, dwin, col, half), a1);
          a2 = wmma_f16(qA_i[dwin], ldB_k(s_ki, sub, dwin, col, half), a2);
          b1 = wmma_f16(qA_r[dwin], ldB_k(s_ki, sub, dwin, col, half), b1);
          b1 = wmma_f16(qA_i[dwin], ldB_k(s_kr, sub, dwin, col, half), b1);
        }
        srt[sub] = a1 - a2;   // Sr = QrKr^T - QiKi^T
        sit[sub] = b1;        // Si = QrKi^T + QiKr^T
      }

      const float apk0 = apS[kt * KTILE + col];
      const float apk1 = apS[kt * KTILE + 16 + col];

      // ---- online softmax, C-layout: row = slot(+8*half), col = lane&15 ----
#pragma unroll
      for (int r = 0; r < 8; ++r) {
        float m0 = sqrtf(srt[0][r] * srt[0][r] + sit[0][r] * sit[0][r])
                   * (0.125f * apq[r]) * apk0;
        float m1 = sqrtf(srt[1][r] * srt[1][r] + sit[1][r] * sit[1][r])
                   * (0.125f * apq[r]) * apk1;
        float mx = fmaxf(m0, m1);
#pragma unroll
        for (int off = 8; off > 0; off >>= 1)
          mx = fmaxf(mx, __shfl_xor(mx, off, 16));
        float mn    = fmaxf(mrow[r], mx);
        float alpha = __expf(mrow[r] - mn);
        float p0    = __expf(m0 - mn);
        float p1    = __expf(m1 - mn);
        float ps    = p0 + p1;
#pragma unroll
        for (int off = 8; off > 0; off >>= 1)
          ps += __shfl_xor(ps, off, 16);
        lrow[r] = lrow[r] * alpha + ps;
        mrow[r] = mn;
#pragma unroll
        for (int dt = 0; dt < 4; ++dt) { oR[dt][r] *= alpha; oI[dt][r] *= alpha; }
        // spill P (f16) C-layout -> LDS for A-layout reload
        s_p[w][(r + 8 * half) * KTILE + col]      = (_Float16)p0;
        s_p[w][(r + 8 * half) * KTILE + 16 + col] = (_Float16)p1;
      }
      __syncthreads();

      // ---- PV: O += W(16x32) @ V(32x64), r & i planes (8 WMMAs) ----
      const v16h wA = ldA_p(&s_p[w][0], col, half);
#pragma unroll
      for (int dt = 0; dt < 4; ++dt) {
        oR[dt] = wmma_f16(wA, ldB_v(s_vtr, dt, col, half), oR[dt]);
        oI[dt] = wmma_f16(wA, ldB_v(s_vti, dt, col, half), oI[dt]);
      }
    }

    // ---- 1/l normalize and accumulate into cross-scale LDS (disjoint rows) --
#pragma unroll
    for (int r = 0; r < 8; ++r) {
      float inv = 1.0f / lrow[r];
      int row = w * 16 + r + 8 * half;
#pragma unroll
      for (int dt = 0; dt < 4; ++dt) {
        s_accR[row * DDIM + dt * 16 + col] += oR[dt][r] * inv;
        s_accI[row * DDIM + dt * 16 + col] += oI[dt][r] * inv;
      }
    }
  }
  __syncthreads();

  // ---- epilogue: complex expert multiply, write [2,B,H,L,D] ----
  const size_t plane = (size_t)BDIM * HDIM * LDIM * DDIM;
  for (int i = tid; i < QTILE * DDIM; i += 128) {
    int row = i >> 6, d = i & 63;
    int l = qblk * QTILE + row;
    float ar = s_accR[i], ai = s_accI[i];
    float er = epr[l * DDIM + d], ei = epi[l * DDIM + d];
    size_t o = ((size_t)bh * LDIM + l) * DDIM + d;
    out[o]         = ar * er - ai * ei;
    out[plane + o] = ar * ei + ai * er;
  }
}

// ---------------------------------------------------------------------------
extern "C" void kernel_launch(void* const* d_in, const int* in_sizes, int n_in,
                              void* d_out, int out_size, void* d_ws, size_t ws_size,
                              hipStream_t stream) {
  (void)in_sizes; (void)n_in; (void)out_size; (void)ws_size;
  const float* Qr = (const float*)d_in[0];
  const float* Qi = (const float*)d_in[1];
  const float* Kr = (const float*)d_in[2];
  const float* Ki = (const float*)d_in[3];
  const float* Vr = (const float*)d_in[4];
  const float* Vi = (const float*)d_in[5];
  float* out = (float*)d_out;

  // ws: ap[4*16*1024] | epr[1024*64] | epi[1024*64]  (768 KB total)
  float* ap  = (float*)d_ws;
  float* epr = ap + 4 * HDIM * LDIM;
  float* epi = epr + LDIM * DDIM;

  ap_kernel<<<dim3(64), dim3(256), 0, stream>>>(ap);
  expert_kernel<<<dim3((LDIM * DDIM) / 256), dim3(256), 0, stream>>>(epr, epi);

  dim3 grid(LDIM / QTILE, BDIM * HDIM);
  attn_kernel<<<grid, dim3(128), 0, stream>>>(Qr, Qi, Kr, Ki, Vr, Vi,
                                              ap, epr, epi, out);
}
